// TransformerBlock_85993835200663
// MI455X (gfx1250) — matrix operand order repaired and checked by reading, not compiled, not run on hardware
//
#include <hip/hip_runtime.h>

// ---------------- problem constants ----------------
constexpr int BB = 4;         // batch
constexpr int NN = 2048;      // sequence
constexpr int EE = 512;       // embed
constexpr int HH = 8;         // heads
constexpr int HD = 64;        // head dim
constexpr int FF = 2048;      // hidden
constexpr int MM = BB * NN;   // 8192 rows

typedef __attribute__((ext_vector_type(16))) __bf16         v16bf;
typedef __attribute__((ext_vector_type(8)))  float          v8f;
typedef __attribute__((ext_vector_type(8)))  unsigned short v8u;
typedef __attribute__((ext_vector_type(16))) unsigned short v16u;
typedef int v4i __attribute__((vector_size(4 * sizeof(int))));

// ---------------- scalar helpers ----------------
__device__ __forceinline__ unsigned short f2bf(float f) {
  union { float f; unsigned u; } c; c.f = f;
  unsigned u = c.u;
  u += 0x7FFFu + ((u >> 16) & 1u);   // round-to-nearest-even
  return (unsigned short)(u >> 16);
}
__device__ __forceinline__ float bf2f(unsigned short h) {
  union { unsigned u; float f; } c; c.u = ((unsigned)h) << 16;
  return c.f;
}
__device__ __forceinline__ v8f zero8() {
  v8f z;
#pragma unroll
  for (int i = 0; i < 8; ++i) z[i] = 0.0f;
  return z;
}

// D = A*B + C, bf16 inputs, f32 accum (v_wmma_f32_16x16x32_bf16)
__device__ __forceinline__ v8f wmma_bf16(v16bf a, v16bf b, v8f c) {
  return __builtin_amdgcn_wmma_f32_16x16x32_bf16(false, a, false, b,
                                                 (short)0, c, false, false);
}

// ---------------- async global->LDS (CDNA5) with fallback ----------------
#if __has_builtin(__builtin_amdgcn_global_load_async_to_lds_b128)
#define USE_ASYNC 1
__device__ __forceinline__ void async_b128(void* lds, const void* g) {
  __builtin_amdgcn_global_load_async_to_lds_b128(
      (__attribute__((address_space(1))) v4i*)g,
      (__attribute__((address_space(3))) v4i*)lds, 0, 0);
}
__device__ __forceinline__ void wait_async(int n) {
#if __has_builtin(__builtin_amdgcn_s_wait_asynccnt)
  if (n == 0) __builtin_amdgcn_s_wait_asynccnt(0);
  else        __builtin_amdgcn_s_wait_asynccnt(8);
#else
  if (n == 0) asm volatile("s_wait_asynccnt 0" ::: "memory");
  else        asm volatile("s_wait_asynccnt 8" ::: "memory");
#endif
}
#else
#define USE_ASYNC 0
__device__ __forceinline__ void wait_async(int) {}
#endif

// ---------------- fragment loaders (CDNA5 16-bit layouts) ----------------
// A-matrix 16x32 from row-major [M][K] (ld elems).  Lane L(0..15): M=L,
// K=0..7 & 16..23 ; lane L+16: same M, K=8..15 & 24..31.  Two 16B loads.
__device__ __forceinline__ v16bf frag_a(const unsigned short* p, int ld) {
  int lane = threadIdx.x & 31;
  int m    = lane & 15;
  int kb   = (lane & 16) ? 8 : 0;
  const unsigned short* r = p + (size_t)m * ld + kb;
  v8u lo = *(const v8u*)(r);
  v8u hi = *(const v8u*)(r + 16);
  v16u t = __builtin_shufflevector(lo, hi, 0, 1, 2, 3, 4, 5, 6, 7,
                                           8, 9, 10, 11, 12, 13, 14, 15);
  return __builtin_bit_cast(v16bf, t);
}

// B-matrix 32x16 from [N][K] row-major storage (B-transposed; ld elems).
// Lanes 0..15 hold K=0..7 & 16..23, lanes 16..31 hold K=8..15 & 24..31.  Two 16B loads.
__device__ __forceinline__ v16bf frag_b(const unsigned short* p, int ld) {
  int lane = threadIdx.x & 31;
  int n    = lane & 15;
  int kb   = (lane & 16) ? 8 : 0;
  const unsigned short* r = p + (size_t)n * ld + kb;
  v8u lo = *(const v8u*)(r);
  v8u hi = *(const v8u*)(r + 16);
  v16u t = __builtin_shufflevector(lo, hi, 0, 1, 2, 3, 4, 5, 6, 7,
                                           8, 9, 10, 11, 12, 13, 14, 15);
  return __builtin_bit_cast(v16bf, t);
}

// ---------------- cast kernels ----------------
__global__ __launch_bounds__(256) void cast_f32_bf16(const float* __restrict__ in,
                                                     unsigned short* __restrict__ out,
                                                     int n) {
  int i = blockIdx.x * 256 + threadIdx.x;
  if (i < n) out[i] = f2bf(in[i]);
}

// W[K][N] f32  ->  Wt[N][K] bf16
__global__ __launch_bounds__(256) void cast_transpose(const float* __restrict__ in,
                                                      unsigned short* __restrict__ out,
                                                      int K, int N) {
  int i = blockIdx.x * 256 + threadIdx.x;
  if (i < K * N) {
    int k = i / N, n = i - k * N;
    out[(size_t)n * K + k] = f2bf(in[i]);
  }
}

// ---------------- tiled bf16 GEMM ----------------
// C[M,N] = A[M,K] * B[K,N] ; A row-major [M][K], Bt pre-transposed [N][K].
// Block 256 = 8 waves ; tile BM=128, BN=128, BK=64 ; wave = M-tile.
// Double-buffered LDS, async global->LDS staging when available.
// mode: 0 = f32 out, 1 = bf16 out, 2 = bf16 out scattered as V^T [b,h,d,n].
__global__ __launch_bounds__(256) void gemm_bf16(const unsigned short* __restrict__ A,
                                                 const unsigned short* __restrict__ Bt,
                                                 void* __restrict__ C,
                                                 int M, int N, int K, int mode) {
  __shared__ __align__(32) unsigned short As[2][128 * 64];   // 2 x 16 KB
  __shared__ __align__(32) unsigned short Bs[2][128 * 64];   // 2 x 16 KB
  int t    = threadIdx.x;
  int wave = t >> 5;
  int lane = t & 31;
  int m0   = blockIdx.x * 128;
  int n0   = blockIdx.y * 128;

  v8f acc[8];
#pragma unroll
  for (int nt = 0; nt < 8; ++nt) acc[nt] = zero8();

  // staging: each thread owns 32 contiguous halves (64 B) of one row per array
  int srow = t >> 1, soff = (t & 1) * 32;
  const unsigned short* ga = A  + (size_t)(m0 + srow) * K + soff;
  const unsigned short* gb = Bt + (size_t)(n0 + srow) * K + soff;

  auto stage = [&](int buf, int kk) {
    unsigned short* da = &As[buf][srow * 64 + soff];
    unsigned short* db = &Bs[buf][srow * 64 + soff];
    const unsigned short* sa = ga + kk;
    const unsigned short* sb = gb + kk;
#if USE_ASYNC
#pragma unroll
    for (int j = 0; j < 4; ++j) {
      async_b128(da + j * 8, sa + j * 8);
      async_b128(db + j * 8, sb + j * 8);
    }
#else
    ((v16u*)da)[0] = ((const v16u*)sa)[0];
    ((v16u*)da)[1] = ((const v16u*)sa)[1];
    ((v16u*)db)[0] = ((const v16u*)sb)[0];
    ((v16u*)db)[1] = ((const v16u*)sb)[1];
#endif
  };

  stage(0, 0);

  for (int k0 = 0; k0 < K; k0 += 64) {
    int buf = (k0 >> 6) & 1;
    bool more = (k0 + 64) < K;
    if (more) stage(buf ^ 1, k0 + 64);          // prefetch next buffer (async)
    wait_async(more ? 8 : 0);                   // prior-buffer copies landed
    __syncthreads();                            // ...for every thread

    // hoist all fragments, then issue the WMMA burst
    v16bf af0 = frag_a(&As[buf][wave * 16 * 64], 64);
    v16bf af1 = frag_a(&As[buf][wave * 16 * 64 + 32], 64);
    v16bf bfr[16];
#pragma unroll
    for (int nt = 0; nt < 8; ++nt) {
      bfr[2 * nt]     = frag_b(&Bs[buf][nt * 16 * 64], 64);
      bfr[2 * nt + 1] = frag_b(&Bs[buf][nt * 16 * 64 + 32], 64);
    }
#pragma unroll
    for (int nt = 0; nt < 8; ++nt) {
      acc[nt] = wmma_bf16(af0, bfr[2 * nt],     acc[nt]);
      acc[nt] = wmma_bf16(af1, bfr[2 * nt + 1], acc[nt]);
    }
    __syncthreads();                            // reads done before re-stage
  }

  int col   = lane & 15;
  int rbase = (lane & 16) ? 8 : 0;
#pragma unroll
  for (int nt = 0; nt < 8; ++nt) {
#pragma unroll
    for (int r = 0; r < 8; ++r) {
      float v    = acc[nt][r];
      size_t row = (size_t)(m0 + wave * 16 + rbase + r);
      int    c   = n0 + nt * 16 + col;
      if (mode == 0) {
        ((float*)C)[row * N + c] = v;
      } else if (mode == 1) {
        ((unsigned short*)C)[row * N + c] = f2bf(v);
      } else {   // V^T: row=(b,nseq), c=(h,d) -> [(b*H+h)*HD+d][nseq]
        int b = (int)(row >> 11), ns = (int)(row & 2047);
        int h = c >> 6, d = c & 63;
        ((unsigned short*)C)[((size_t)(b * HH + h) * HD + d) * NN + ns] = f2bf(v);
      }
    }
  }
}

// ---------------- eps * colsum(V) over Vt [b,h,d,n] ----------------
__global__ __launch_bounds__(256) void colsum_v(const unsigned short* __restrict__ Vt,
                                                float* __restrict__ SV) {
  int i = blockIdx.x * 256 + threadIdx.x;   // 0..2047 == (b*H+h)*HD+d
  const unsigned short* p = Vt + (size_t)i * NN;
  float s = 0.f;
  for (int k = 0; k < NN; ++k) s += bf2f(p[k]);
  SV[i] = s;
}

// ---------------- attention pass 1: softmax stats ----------------
// grid (NN/32, HH, BB), 256 threads. Wave w handles k in {w*32 + s*256}.
__global__ __launch_bounds__(256) void attn_stats(const unsigned short* __restrict__ Qf,
                                                  const unsigned short* __restrict__ Kf,
                                                  float* __restrict__ Sm,
                                                  float* __restrict__ Szi) {
  int qt = blockIdx.x, h = blockIdx.y, b = blockIdx.z;
  int t = threadIdx.x, wave = t >> 5, lane = t & 31;
  int q0 = qt * 32;
  const unsigned short* Qbase = Qf + (size_t)(b * NN + q0) * EE + h * HD;

  v16bf qa[2][2];
#pragma unroll
  for (int mt = 0; mt < 2; ++mt)
#pragma unroll
    for (int ks = 0; ks < 2; ++ks)
      qa[mt][ks] = frag_a(Qbase + mt * 16 * EE + ks * 32, EE);

  float mrun[16], zrun[16];
#pragma unroll
  for (int i = 0; i < 16; ++i) { mrun[i] = -1e30f; zrun[i] = 0.f; }

  for (int k0 = wave * 32; k0 < NN; k0 += 256) {
    const unsigned short* Kbase = Kf + (size_t)(b * NN + k0) * EE + h * HD;
    v16bf kb_[2][2];
#pragma unroll
    for (int nt = 0; nt < 2; ++nt)
#pragma unroll
      for (int ks = 0; ks < 2; ++ks)
        kb_[nt][ks] = frag_b(Kbase + nt * 16 * EE + ks * 32, EE);
#pragma unroll
    for (int mt = 0; mt < 2; ++mt)
#pragma unroll
      for (int nt = 0; nt < 2; ++nt) {
        v8f s = zero8();
        s = wmma_bf16(qa[mt][0], kb_[nt][0], s);
        s = wmma_bf16(qa[mt][1], kb_[nt][1], s);
#pragma unroll
        for (int r = 0; r < 8; ++r) {
          float v  = s[r] * 8.0f;                 // scores / SCALE, SCALE = 1/8
          int   i  = mt * 8 + r;
          float mn = fmaxf(mrun[i], v);
          zrun[i]  = zrun[i] * __expf(mrun[i] - mn) + __expf(v - mn);
          mrun[i]  = mn;
        }
      }
  }

  // reduce across the 16 lanes sharing each row (xor masks stay in half-wave)
#pragma unroll
  for (int off = 1; off < 16; off <<= 1) {
#pragma unroll
    for (int i = 0; i < 16; ++i) {
      float mo = __shfl_xor(mrun[i], off, 32);
      float zo = __shfl_xor(zrun[i], off, 32);
      float mn = fmaxf(mrun[i], mo);
      zrun[i]  = zrun[i] * __expf(mrun[i] - mn) + zo * __expf(mo - mn);
      mrun[i]  = mn;
    }
  }

  __shared__ float sm[8][32];
  __shared__ float sz[8][32];
  if ((lane & 15) == 0) {
    int rbase = (lane & 16) ? 8 : 0;
#pragma unroll
    for (int i = 0; i < 16; ++i) {
      int row = (i >> 3) * 16 + rbase + (i & 7);
      sm[wave][row] = mrun[i];
      sz[wave][row] = zrun[i];
    }
  }
  __syncthreads();
  if (t < 32) {
    float m = -1e30f, z = 0.f;
    for (int w = 0; w < 8; ++w) {
      float mo = sm[w][t], zo = sz[w][t];
      float mn = fmaxf(m, mo);
      z = z * __expf(m - mn) + zo * __expf(mo - mn);
      m = mn;
    }
    size_t idx = (size_t)(b * HH + h) * NN + q0 + t;
    Sm[idx]  = m;
    Szi[idx] = 1.0f / z;
  }
}

// ---------------- attention pass 2 ----------------
// grid (NN/32, BB), 256 threads; wave = head.  V comes pre-transposed [b,h,d,n].
__global__ __launch_bounds__(256) void attn_pass2(const unsigned short* __restrict__ Qf,
                                                  const unsigned short* __restrict__ Kf,
                                                  const unsigned short* __restrict__ Vt,
                                                  const float* __restrict__ Sm,
                                                  const float* __restrict__ Szi,
                                                  const float* __restrict__ SV,
                                                  unsigned short* __restrict__ Of) {
  __shared__ __align__(32) unsigned short Ps[HH][32 * 32];   // 16 KB
  __shared__ float Dinv[32 * 32];                            // 4 KB
  int qt = blockIdx.x, b = blockIdx.y;
  int t = threadIdx.x, h = t >> 5, lane = t & 31;
  int q0 = qt * 32;
  int col = lane & 15, rbase = (lane & 16) ? 8 : 0;

  const unsigned short* Qbase = Qf + (size_t)(b * NN + q0) * EE + h * HD;
  v16bf qa[2][2];
#pragma unroll
  for (int mt = 0; mt < 2; ++mt)
#pragma unroll
    for (int ks = 0; ks < 2; ++ks)
      qa[mt][ks] = frag_a(Qbase + mt * 16 * EE + ks * 32, EE);

  float mrow[16], zirow[16];
#pragma unroll
  for (int i = 0; i < 16; ++i) {
    int row = (i >> 3) * 16 + rbase + (i & 7);
    size_t sidx = (size_t)(b * HH + h) * NN + q0 + row;
    mrow[i]  = Sm[sidx];
    zirow[i] = Szi[sidx];
  }

  v8f oacc[2][4];
#pragma unroll
  for (int mt = 0; mt < 2; ++mt)
#pragma unroll
    for (int nt = 0; nt < 4; ++nt) oacc[mt][nt] = zero8();

  const unsigned short* Vhead = Vt + (size_t)(b * HH + h) * HD * NN;

  for (int k0 = 0; k0 < NN; k0 += 32) {
    // per-head scores -> softmax numerator with pass-1 stats
    const unsigned short* Kbase = Kf + (size_t)(b * NN + k0) * EE + h * HD;
    v16bf kb_[2][2];
#pragma unroll
    for (int nt = 0; nt < 2; ++nt)
#pragma unroll
      for (int ks = 0; ks < 2; ++ks)
        kb_[nt][ks] = frag_b(Kbase + nt * 16 * EE + ks * 32, EE);

    float aval[2][2][8];
#pragma unroll
    for (int mt = 0; mt < 2; ++mt)
#pragma unroll
      for (int nt = 0; nt < 2; ++nt) {
        v8f s = zero8();
        s = wmma_bf16(qa[mt][0], kb_[nt][0], s);
        s = wmma_bf16(qa[mt][1], kb_[nt][1], s);
#pragma unroll
        for (int r = 0; r < 8; ++r) {
          int i = mt * 8 + r;
          aval[mt][nt][r] = __expf(s[r] * 8.0f - mrow[i]) * zirow[i];
        }
      }

#pragma unroll
    for (int mt = 0; mt < 2; ++mt)
#pragma unroll
      for (int nt = 0; nt < 2; ++nt)
#pragma unroll
        for (int r = 0; r < 8; ++r)
          Ps[h][(mt * 16 + rbase + r) * 32 + nt * 16 + col] = f2bf(aval[mt][nt][r]);
    __syncthreads();

    // cross-head denominator D(q,k)
#pragma unroll
    for (int j = 0; j < 4; ++j) {
      int idx = t * 4 + j;
      float s = 0.f;
#pragma unroll
      for (int hh = 0; hh < HH; ++hh) s += bf2f(Ps[hh][idx]);
      Dinv[idx] = 1.0f / s;
    }
    __syncthreads();

    // P' = P / D (wave-local region; per-wave DS ordering covers the reload)
#pragma unroll
    for (int mt = 0; mt < 2; ++mt)
#pragma unroll
      for (int nt = 0; nt < 2; ++nt)
#pragma unroll
        for (int r = 0; r < 8; ++r) {
          int pidx = (mt * 16 + rbase + r) * 32 + nt * 16 + col;
          Ps[h][pidx] = f2bf(aval[mt][nt][r] * Dinv[pidx]);
        }

    v16bf pa0 = frag_a(&Ps[h][0], 32);
    v16bf pa1 = frag_a(&Ps[h][16 * 32], 32);
#pragma unroll
    for (int nt = 0; nt < 4; ++nt) {
      v16bf vb = frag_b(Vhead + (size_t)(nt * 16) * NN + k0, NN);
      oacc[0][nt] = wmma_bf16(pa0, vb, oacc[0][nt]);
      oacc[1][nt] = wmma_bf16(pa1, vb, oacc[1][nt]);
    }
  }

  // += eps * colsum(V) ; write out [b, q, h, d]
#pragma unroll
  for (int mt = 0; mt < 2; ++mt)
#pragma unroll
    for (int nt = 0; nt < 4; ++nt) {
      float sv = 1e-6f * SV[(b * HH + h) * HD + nt * 16 + col];
#pragma unroll
      for (int r = 0; r < 8; ++r) {
        int row = q0 + mt * 16 + rbase + r;
        Of[(size_t)(b * NN + row) * EE + h * HD + nt * 16 + col] =
            f2bf(oacc[mt][nt][r] + sv);
      }
    }
}

// ---------------- layernorm kernels ----------------
__device__ __forceinline__ void block_sum2(float a, float b, float* out2,
                                           float* scratch) {
  int t = threadIdx.x;
#pragma unroll
  for (int off = 16; off; off >>= 1) {
    a += __shfl_xor(a, off, 32);
    b += __shfl_xor(b, off, 32);
  }
  if ((t & 31) == 0) { scratch[(t >> 5) * 2] = a; scratch[(t >> 5) * 2 + 1] = b; }
  __syncthreads();
  if (t == 0) {
    float sa = 0.f, sb = 0.f;
    for (int w = 0; w < 8; ++w) { sa += scratch[w * 2]; sb += scratch[w * 2 + 1]; }
    out2[0] = sa; out2[1] = sb;
  }
  __syncthreads();
}

// out = (base ? base + : ) (relu?) LN(src)   for rows of 512 f32
__global__ __launch_bounds__(256) void ln_row512(const float* __restrict__ src,
                                                 const float* __restrict__ base,
                                                 const float* __restrict__ g,
                                                 const float* __restrict__ be,
                                                 float* __restrict__ outf,
                                                 unsigned short* __restrict__ outb,
                                                 int do_relu) {
  __shared__ float scratch[16];
  __shared__ float out2[2];
  size_t row = blockIdx.x;
  int t = threadIdx.x;
  const float* p = src + row * EE;
  float v0 = p[t], v1 = p[t + 256];
  block_sum2(v0 + v1, v0 * v0 + v1 * v1, out2, scratch);
  float mu   = out2[0] * (1.0f / EE);
  float var  = out2[1] * (1.0f / EE) - mu * mu;
  float rstd = rsqrtf(var + 1e-5f);
  float y0 = (v0 - mu) * rstd * g[t] + be[t];
  float y1 = (v1 - mu) * rstd * g[t + 256] + be[t + 256];
  if (do_relu) { y0 = fmaxf(y0, 0.f); y1 = fmaxf(y1, 0.f); }
  if (base)    { y0 += base[row * EE + t]; y1 += base[row * EE + t + 256]; }
  outf[row * EE + t]       = y0;
  outf[row * EE + t + 256] = y1;
  if (outb) {
    outb[row * EE + t]       = f2bf(y0);
    outb[row * EE + t + 256] = f2bf(y1);
  }
}

// in-place relu(LN(row)) for rows of 2048 bf16
__global__ __launch_bounds__(256) void ln_relu_row2048_bf16(
    unsigned short* __restrict__ data, const float* __restrict__ g,
    const float* __restrict__ be) {
  __shared__ float scratch[16];
  __shared__ float out2[2];
  size_t row = blockIdx.x;
  int t = threadIdx.x;
  unsigned short* p = data + row * FF;
  float v[8], s = 0.f, s2 = 0.f;
#pragma unroll
  for (int j = 0; j < 8; ++j) {
    v[j] = bf2f(p[t + j * 256]);
    s += v[j];
    s2 += v[j] * v[j];
  }
  block_sum2(s, s2, out2, scratch);
  float mu   = out2[0] * (1.0f / FF);
  float var  = out2[1] * (1.0f / FF) - mu * mu;
  float rstd = rsqrtf(var + 1e-5f);
#pragma unroll
  for (int j = 0; j < 8; ++j) {
    int c = t + j * 256;
    float y = fmaxf((v[j] - mu) * rstd * g[c] + be[c], 0.f);
    p[c] = f2bf(y);
  }
}

// ---------------- host launch ----------------
extern "C" void kernel_launch(void* const* d_in, const int* in_sizes, int n_in,
                              void* d_out, int out_size, void* d_ws, size_t ws_size,
                              hipStream_t stream) {
  (void)in_sizes; (void)n_in; (void)out_size; (void)ws_size;
  const float* x   = (const float*)d_in[0];
  const float* WQ  = (const float*)d_in[1];
  const float* WK  = (const float*)d_in[2];
  const float* WV  = (const float*)d_in[3];
  const float* WO  = (const float*)d_in[4];
  const float* g1  = (const float*)d_in[5];
  const float* b1  = (const float*)d_in[6];
  const float* g2  = (const float*)d_in[7];
  const float* b2  = (const float*)d_in[8];
  const float* Wf1 = (const float*)d_in[9];
  const float* gf1 = (const float*)d_in[10];
  const float* bf1 = (const float*)d_in[11];
  const float* Wf2 = (const float*)d_in[12];
  const float* gf2 = (const float*)d_in[13];
  const float* bf2 = (const float*)d_in[14];

  char* ws = (char*)d_ws;
  size_t off = 0;
  auto alloc = [&](size_t bytes) -> char* {
    char* p = ws + off;
    off += (bytes + 255) & ~((size_t)255);
    return p;
  };
  unsigned short* xb   = (unsigned short*)alloc((size_t)MM * EE * 2);
  unsigned short* wqT  = (unsigned short*)alloc((size_t)EE * EE * 2);
  unsigned short* wkT  = (unsigned short*)alloc((size_t)EE * EE * 2);
  unsigned short* wvT  = (unsigned short*)alloc((size_t)EE * EE * 2);
  unsigned short* woT  = (unsigned short*)alloc((size_t)EE * EE * 2);
  unsigned short* wf1T = (unsigned short*)alloc((size_t)EE * FF * 2);
  unsigned short* wf2T = (unsigned short*)alloc((size_t)FF * EE * 2);
  unsigned short* Qf   = (unsigned short*)alloc((size_t)MM * EE * 2);
  unsigned short* Kf   = (unsigned short*)alloc((size_t)MM * EE * 2);
  unsigned short* Vt   = (unsigned short*)alloc((size_t)MM * EE * 2);
  float*          SVb  = (float*)alloc((size_t)BB * HH * HD * 4);
  float*          Smb  = (float*)alloc((size_t)BB * HH * NN * 4);
  float*          Szib = (float*)alloc((size_t)BB * HH * NN * 4);
  unsigned short* Ofb  = (unsigned short*)alloc((size_t)MM * EE * 2);
  float*          proj = (float*)alloc((size_t)MM * EE * 4);
  float*          y1   = (float*)alloc((size_t)MM * EE * 4);
  unsigned short* y1b  = (unsigned short*)alloc((size_t)MM * EE * 2);
  unsigned short* h1   = (unsigned short*)alloc((size_t)MM * FF * 2);
  float*          h2   = (float*)alloc((size_t)MM * EE * 4);

  // casts: x straight, weights transposed to [N][K]
  cast_f32_bf16<<<(MM * EE) / 256, 256, 0, stream>>>(x, xb, MM * EE);
  cast_transpose<<<(EE * EE) / 256, 256, 0, stream>>>(WQ,  wqT,  EE, EE);
  cast_transpose<<<(EE * EE) / 256, 256, 0, stream>>>(WK,  wkT,  EE, EE);
  cast_transpose<<<(EE * EE) / 256, 256, 0, stream>>>(WV,  wvT,  EE, EE);
  cast_transpose<<<(EE * EE) / 256, 256, 0, stream>>>(WO,  woT,  EE, EE);
  cast_transpose<<<(EE * FF) / 256, 256, 0, stream>>>(Wf1, wf1T, EE, FF);
  cast_transpose<<<(FF * EE) / 256, 256, 0, stream>>>(Wf2, wf2T, FF, EE);

  // Q, K ([b,n,h*d] bf16) ; V written transposed [b,h,d,n]
  gemm_bf16<<<dim3(MM / 128, EE / 128), 256, 0, stream>>>(xb, wqT, Qf, MM, EE, EE, 1);
  gemm_bf16<<<dim3(MM / 128, EE / 128), 256, 0, stream>>>(xb, wkT, Kf, MM, EE, EE, 1);
  gemm_bf16<<<dim3(MM / 128, EE / 128), 256, 0, stream>>>(xb, wvT, Vt, MM, EE, EE, 2);

  colsum_v<<<(BB * HH * HD) / 256, 256, 0, stream>>>(Vt, SVb);
  attn_stats<<<dim3(NN / 32, HH, BB), 256, 0, stream>>>(Qf, Kf, Smb, Szib);
  attn_pass2<<<dim3(NN / 32, BB), 256, 0, stream>>>(Qf, Kf, Vt, Smb, Szib, SVb, Ofb);

  // output projection + first residual/LN
  gemm_bf16<<<dim3(MM / 128, EE / 128), 256, 0, stream>>>(Ofb, woT, proj, MM, EE, EE, 0);
  ln_row512<<<MM, 256, 0, stream>>>(proj, x, g1, b1, y1, y1b, 0);

  // FFN
  gemm_bf16<<<dim3(MM / 128, FF / 128), 256, 0, stream>>>(y1b, wf1T, h1, MM, FF, EE, 1);
  ln_relu_row2048_bf16<<<MM, 256, 0, stream>>>(h1, gf1, bf1);
  gemm_bf16<<<dim3(MM / 128, EE / 128), 256, 0, stream>>>(h1, wf2T, h2, MM, EE, FF, 0);
  ln_row512<<<MM, 256, 0, stream>>>(h2, nullptr, gf2, bf2, h2, nullptr, 1);

  // final residual + LN -> d_out (f32)
  ln_row512<<<MM, 256, 0, stream>>>(h2, y1, g2, b2, (float*)d_out, nullptr, 0);
}
